// GlobalFeatureGAT_17815524343963
// MI455X (gfx1250) — matrix operand order repaired and checked by reading, not compiled, not run on hardware
//
#include <hip/hip_runtime.h>

// ---------------- types for WMMA ----------------
typedef __bf16 bf16_t;
typedef __attribute__((ext_vector_type(16))) __bf16 v16bf;
typedef __attribute__((ext_vector_type(8)))  __bf16 v8bf;
typedef __attribute__((ext_vector_type(8)))  float  v8f;

#define SF   16
#define ED   32
#define LTOK 20
#define DIN  176
#define KP1  192      // 176 padded to a multiple of 32 for the WMMA K loop
#define HC   128
#define NH   4
#define H1C  512      // NH*HC
#define NB   8

// float -> bf16 (round to nearest even), via bits; avoids relying on __bf16 arithmetic
__device__ __forceinline__ unsigned short f2bf(float f) {
  unsigned u = __float_as_uint(f);
  u += 0x7FFFu + ((u >> 16) & 1u);
  return (unsigned short)(u >> 16);
}

// monotonic encoding of float into uint so atomicMax == float max
__device__ __forceinline__ unsigned encf(float f) {
  unsigned u = __float_as_uint(f);
  return (u & 0x80000000u) ? ~u : (u | 0x80000000u);
}
__device__ __forceinline__ float decf(unsigned u) {
  u = (u & 0x80000000u) ? (u & 0x7FFFFFFFu) : ~u;
  return __uint_as_float(u);
}

__device__ __forceinline__ void edge_sd(const int* __restrict__ ei, int E, int e,
                                        int& s, int& d) {
  if (e < E) { s = ei[e]; d = ei[E + e]; }       // edge_index rows: [0]=src, [1]=dst
  else       { s = d = e - E; }                  // appended self-loops
}

// ---------------- K1: embeddings masked-mean + concat ----------------
__global__ void k_embed_concat(const float* __restrict__ xs,
                               const int* xi0, const int* xi1, const int* xi2,
                               const int* xi3, const int* xi4,
                               const float* t0, const float* t1, const float* t2,
                               const float* t3, const float* t4,
                               float* __restrict__ feats, int N) {
  int wave = (blockIdx.x * blockDim.x + threadIdx.x) >> 5;
  int lane = threadIdx.x & 31;
  if (wave >= N) return;
  const int n = wave;
  if (lane < SF) feats[(size_t)n * DIN + lane] = xs[(size_t)n * SF + lane];
  const int*   xis[5] = {xi0, xi1, xi2, xi3, xi4};
  const float* tbs[5] = {t0, t1, t2, t3, t4};
#pragma unroll
  for (int f = 0; f < 5; ++f) {
    const int* ip = xis[f] + (size_t)n * LTOK;
    const float* tab = tbs[f];
    float acc = 0.f, cnt = 0.f;
#pragma unroll
    for (int t = 0; t < LTOK; ++t) {
      int id = ip[t];
      if (id != 0) { acc += tab[(size_t)id * ED + lane]; cnt += 1.f; }
    }
    feats[(size_t)n * DIN + SF + f * ED + lane] = acc / (cnt + 1e-9f);
  }
}

// ---------------- K2: LayerNorm + bf16 + pad K to 192 ----------------
__global__ void k_layernorm_bf16(const float* __restrict__ x,
                                 const float* __restrict__ g, const float* __restrict__ b,
                                 unsigned short* __restrict__ A, int N) {
  int wave = (blockIdx.x * blockDim.x + threadIdx.x) >> 5;
  int lane = threadIdx.x & 31;
  if (wave >= N) return;
  const float* xp = x + (size_t)wave * DIN;
  float v[6]; float s = 0.f;
#pragma unroll
  for (int j = 0; j < 6; ++j) { int i = lane + 32 * j; v[j] = (i < DIN) ? xp[i] : 0.f; s += v[j]; }
  for (int o = 16; o; o >>= 1) s += __shfl_xor(s, o, 32);
  float mu = s * (1.f / DIN);
  float vs = 0.f;
#pragma unroll
  for (int j = 0; j < 6; ++j) { int i = lane + 32 * j; float d = (i < DIN) ? (v[j] - mu) : 0.f; vs += d * d; }
  for (int o = 16; o; o >>= 1) vs += __shfl_xor(vs, o, 32);
  float rstd = rsqrtf(vs * (1.f / DIN) + 1e-5f);
  unsigned short* ap = A + (size_t)wave * KP1;
#pragma unroll
  for (int j = 0; j < 6; ++j) {
    int i = lane + 32 * j;
    ap[i] = (i < DIN) ? f2bf((v[j] - mu) * rstd * g[i] + b[i]) : (unsigned short)0;
  }
}

// ---------------- K3: weight transpose + bf16 (B operand wants rows = out-channel) ----
__global__ void k_w1t(const float* __restrict__ W, unsigned short* __restrict__ Wt) {
  int i = blockIdx.x * blockDim.x + threadIdx.x;             // over 512*192
  if (i >= H1C * KP1) return;
  int n = i / KP1, k = i % KP1;
  Wt[i] = (k < DIN) ? f2bf(W[(size_t)k * H1C + n]) : (unsigned short)0;
}
__global__ void k_w2t(const float* __restrict__ W, unsigned short* __restrict__ Wt) {
  int i = blockIdx.x * blockDim.x + threadIdx.x;             // over 128*512
  if (i >= HC * H1C) return;
  int n = i / H1C, k = i % H1C;
  Wt[i] = f2bf(W[(size_t)k * HC + n]);
}

// ---------------- K4: GEMM, one wave per 16x16 tile, v_wmma_f32_16x16x32_bf16 ------
// A: [Mtiles*16, Kpad] bf16 row-major; Bt: [Ntiles*16, Kpad] bf16 (transposed weights)
__global__ void k_gemm_wmma(const unsigned short* __restrict__ A,
                            const unsigned short* __restrict__ Bt,
                            float* __restrict__ C, int Mtiles, int Ntiles, int Kpad) {
  int wave = (blockIdx.x * blockDim.x + threadIdx.x) >> 5;
  int lane = threadIdx.x & 31;
  if (wave >= Mtiles * Ntiles) return;                       // wave-uniform: EXEC all-1s inside
  int mt = wave / Ntiles, nt = wave % Ntiles;
  // A fragment (ISA 16-bit A 16x32): lanes 0-15 -> K {0..7,16..23}; lanes 16-31 -> K {8..15,24..31}
  int arow = mt * 16 + (lane & 15);
  int ka0  = (lane < 16) ? 0 : 8;
  // B fragment (32x16, K x N): lane&15 = column; lanes 0-15 K {0..7,16..23}, lanes 16-31 K {8..15,24..31}
  int bcol = nt * 16 + (lane & 15);
  int kb0  = (lane < 16) ? 0 : 8;
  const unsigned short* Ap = A  + (size_t)arow * Kpad;
  const unsigned short* Bp = Bt + (size_t)bcol * Kpad;
  v8f acc = {};
  int ktiles = Kpad >> 5;
  for (int kt = 0; kt < ktiles; ++kt) {
    int kb = kt * 32;
    union { v16bf v; v8bf h[2]; } a, b;
    a.h[0] = *(const v8bf*)(Ap + kb + ka0);
    a.h[1] = *(const v8bf*)(Ap + kb + ka0 + 16);
    b.h[0] = *(const v8bf*)(Bp + kb + kb0);
    b.h[1] = *(const v8bf*)(Bp + kb + kb0 + 16);
    acc = __builtin_amdgcn_wmma_f32_16x16x32_bf16(false, a.v, false, b.v,
                                                  (short)0, acc, false, false);
  }
  // C/D layout: VGPR j -> row j (lanes 0-15) / j+8 (lanes 16-31), col = lane&15
  int crow = mt * 16 + ((lane < 16) ? 0 : 8);
  int ccol = nt * 16 + (lane & 15);
  int Nout = Ntiles * 16;
  float* Cp = C + (size_t)crow * Nout + ccol;
#pragma unroll
  for (int j = 0; j < 8; ++j) Cp[(size_t)j * Nout] = acc[j];
}

// ---------------- K5: attention logits al_s/al_d (wave per node) -----------------
__global__ void k_att_prep(const float* __restrict__ h, const float* __restrict__ asrc,
                           const float* __restrict__ adst, float* __restrict__ als,
                           float* __restrict__ ald, int N, int H) {
  int wave = (blockIdx.x * blockDim.x + threadIdx.x) >> 5;
  int lane = threadIdx.x & 31;
  if (wave >= N) return;
  const float* hp = h + (size_t)wave * H * HC;
  for (int hh = 0; hh < H; ++hh) {
    float s = 0.f, d = 0.f;
#pragma unroll
    for (int j = 0; j < 4; ++j) {
      int c = hh * HC + lane + 32 * j;
      float v = hp[c];
      s += v * asrc[c];
      d += v * adst[c];
    }
    for (int o = 16; o; o >>= 1) { s += __shfl_xor(s, o, 32); d += __shfl_xor(d, o, 32); }
    if (lane == 0) { als[(size_t)wave * H + hh] = s; ald[(size_t)wave * H + hh] = d; }
  }
}

// ---------------- K6/K7: edge softmax passes ----------------
__global__ void k_edge_max(const int* __restrict__ ei, int E, int Etot, int H,
                           const float* __restrict__ als, const float* __restrict__ ald,
                           unsigned* __restrict__ m) {
  int idx = blockIdx.x * blockDim.x + threadIdx.x;
  if (idx >= Etot * H) return;
  int e = idx / H, hh = idx - e * H;
  int s, d; edge_sd(ei, E, e, s, d);
  float v = als[(size_t)s * H + hh] + ald[(size_t)d * H + hh];
  v = (v > 0.f) ? v : 0.2f * v;                              // leaky_relu(0.2)
  atomicMax(m + (size_t)d * H + hh, encf(v));
}

__global__ void k_edge_den(const int* __restrict__ ei, int E, int Etot, int H,
                           const float* __restrict__ als, const float* __restrict__ ald,
                           const unsigned* __restrict__ m, float* __restrict__ den) {
  int idx = blockIdx.x * blockDim.x + threadIdx.x;
  if (idx >= Etot * H) return;
  int e = idx / H, hh = idx - e * H;
  int s, d; edge_sd(ei, E, e, s, d);
  float v = als[(size_t)s * H + hh] + ald[(size_t)d * H + hh];
  v = (v > 0.f) ? v : 0.2f * v;
  float ex = __expf(v - decf(m[(size_t)d * H + hh]));
  atomicAdd(den + (size_t)d * H + hh, ex);
}

// ---------------- K8: weighted message aggregation (wave per edge) ----------------
__global__ void k_edge_agg(const int* __restrict__ ei, int E, int Etot, int H,
                           const float* __restrict__ als, const float* __restrict__ ald,
                           const unsigned* __restrict__ m, const float* __restrict__ den,
                           const float* __restrict__ h, float* __restrict__ out) {
  int wave = (blockIdx.x * blockDim.x + threadIdx.x) >> 5;
  int lane = threadIdx.x & 31;
  if (wave >= Etot) return;
  int s, d; edge_sd(ei, E, wave, s, d);
  float alpha = 0.f;
  if (lane < H) {
    float v = als[(size_t)s * H + lane] + ald[(size_t)d * H + lane];
    v = (v > 0.f) ? v : 0.2f * v;
    float ex = __expf(v - decf(m[(size_t)d * H + lane]));
    alpha = ex / (den[(size_t)d * H + lane] + 1e-16f);
  }
  const int HCt = H * HC;
  const float* hs = h + (size_t)s * HCt;
  float* op = out + (size_t)d * HCt;
  for (int j = 0; j < HCt / 32; ++j) {
    int c = lane + 32 * j;
    float a = __shfl(alpha, c >> 7, 32);                     // head = c / 128
    atomicAdd(op + c, hs[c] * a);
  }
}

// ---------------- K9: bias + ELU + bf16 (layer-1 output -> layer-2 A) -------------
__global__ void k_bias_elu_bf16(const float* __restrict__ in, const float* __restrict__ bias,
                                unsigned short* __restrict__ out, int tot, int C) {
  for (int i = blockIdx.x * blockDim.x + threadIdx.x; i < tot; i += gridDim.x * blockDim.x) {
    float v = in[i] + bias[i % C];
    v = (v > 0.f) ? v : (__expf(v) - 1.f);                   // elu(alpha=1)
    out[i] = f2bf(v);
  }
}

// ---------------- K10: bias + ELU + global max-pool (encoded atomicMax) -----------
__global__ void k_final_pool(const float* __restrict__ in, const float* __restrict__ bias,
                             const int* __restrict__ batch, unsigned* __restrict__ pool, int N) {
  int tot = N * HC;
  for (int i = blockIdx.x * blockDim.x + threadIdx.x; i < tot; i += gridDim.x * blockDim.x) {
    int n = i >> 7, c = i & (HC - 1);
    float v = in[i] + bias[c];
    v = (v > 0.f) ? v : (__expf(v) - 1.f);
    atomicMax(pool + (size_t)batch[n] * HC + c, encf(v));
  }
}

__global__ void k_pool_decode(const unsigned* __restrict__ pool, float* __restrict__ out, int n) {
  int i = blockIdx.x * blockDim.x + threadIdx.x;
  if (i < n) out[i] = decf(pool[i]);
}

// ---------------- launch ----------------
extern "C" void kernel_launch(void* const* d_in, const int* in_sizes, int n_in,
                              void* d_out, int out_size, void* d_ws, size_t ws_size,
                              hipStream_t stream) {
  const float* xs    = (const float*)d_in[0];
  const int*   xi0   = (const int*)d_in[1];
  const int*   xi1   = (const int*)d_in[2];
  const int*   xi2   = (const int*)d_in[3];
  const int*   xi3   = (const int*)d_in[4];
  const int*   xi4   = (const int*)d_in[5];
  const int*   ei    = (const int*)d_in[6];
  const int*   batch = (const int*)d_in[7];
  const float* t0 = (const float*)d_in[8];
  const float* t1 = (const float*)d_in[9];
  const float* t2 = (const float*)d_in[10];
  const float* t3 = (const float*)d_in[11];
  const float* t4 = (const float*)d_in[12];
  const float* ln_g = (const float*)d_in[13];
  const float* ln_b = (const float*)d_in[14];
  const float* W1   = (const float*)d_in[15];
  const float* as1  = (const float*)d_in[16];
  const float* ad1  = (const float*)d_in[17];
  const float* b1   = (const float*)d_in[18];
  const float* W2   = (const float*)d_in[19];
  const float* as2  = (const float*)d_in[20];
  const float* ad2  = (const float*)d_in[21];
  const float* b2   = (const float*)d_in[22];

  const int N    = in_sizes[0] / SF;       // 20000 (multiple of 16)
  const int E    = in_sizes[6] / 2;        // 320000
  const int Etot = E + N;
  const int Mt   = N / 16;

  char* w = (char*)d_ws;
  auto carve = [&](size_t bytes) -> void* {
    void* p = (void*)w; w += (bytes + 255) & ~(size_t)255; return p;
  };
  float*          feats = (float*)         carve((size_t)N * DIN * 4);
  unsigned short* A1    = (unsigned short*)carve((size_t)N * KP1 * 2);
  unsigned short* W1t   = (unsigned short*)carve((size_t)H1C * KP1 * 2);
  unsigned short* W2t   = (unsigned short*)carve((size_t)HC * H1C * 2);
  float*          h1    = (float*)         carve((size_t)N * H1C * 4);
  float*          als1  = (float*)         carve((size_t)N * NH * 4);
  float*          ald1  = (float*)         carve((size_t)N * NH * 4);
  unsigned*       m1    = (unsigned*)      carve((size_t)N * NH * 4);
  float*          den1  = (float*)         carve((size_t)N * NH * 4);
  float*          out1  = (float*)         carve((size_t)N * H1C * 4);
  unsigned short* A2    = (unsigned short*)carve((size_t)N * H1C * 2);
  float*          h2    = (float*)         carve((size_t)N * HC * 4);
  float*          als2  = (float*)         carve((size_t)N * 4);
  float*          ald2  = (float*)         carve((size_t)N * 4);
  unsigned*       m2    = (unsigned*)      carve((size_t)N * 4);
  float*          den2  = (float*)         carve((size_t)N * 4);
  float*          out2  = (float*)         carve((size_t)N * HC * 4);
  unsigned*       pool  = (unsigned*)      carve((size_t)NB * HC * 4);

  // zero-init accumulators (0u == encf-min sentinel and 0.0f)
  hipMemsetAsync(out1, 0, (size_t)N * H1C * 4, stream);
  hipMemsetAsync(out2, 0, (size_t)N * HC * 4, stream);
  hipMemsetAsync(m1,   0, (size_t)N * NH * 4, stream);
  hipMemsetAsync(den1, 0, (size_t)N * NH * 4, stream);
  hipMemsetAsync(m2,   0, (size_t)N * 4, stream);
  hipMemsetAsync(den2, 0, (size_t)N * 4, stream);
  hipMemsetAsync(pool, 0, (size_t)NB * HC * 4, stream);

  const int BLK = 256;                       // 8 wave32s per block
  // node features
  k_embed_concat<<<(N + 7) / 8, BLK, 0, stream>>>(xs, xi0, xi1, xi2, xi3, xi4,
                                                  t0, t1, t2, t3, t4, feats, N);
  k_layernorm_bf16<<<(N + 7) / 8, BLK, 0, stream>>>(feats, ln_g, ln_b, A1, N);
  k_w1t<<<(H1C * KP1 + BLK - 1) / BLK, BLK, 0, stream>>>(W1, W1t);
  k_w2t<<<(HC * H1C + BLK - 1) / BLK, BLK, 0, stream>>>(W2, W2t);

  // ---- layer 1 ----
  k_gemm_wmma<<<(Mt * (H1C / 16) + 7) / 8, BLK, 0, stream>>>(A1, W1t, h1, Mt, H1C / 16, KP1);
  k_att_prep<<<(N + 7) / 8, BLK, 0, stream>>>(h1, as1, ad1, als1, ald1, N, NH);
  k_edge_max<<<(Etot * NH + BLK - 1) / BLK, BLK, 0, stream>>>(ei, E, Etot, NH, als1, ald1, m1);
  k_edge_den<<<(Etot * NH + BLK - 1) / BLK, BLK, 0, stream>>>(ei, E, Etot, NH, als1, ald1, m1, den1);
  k_edge_agg<<<(Etot + 7) / 8, BLK, 0, stream>>>(ei, E, Etot, NH, als1, ald1, m1, den1, h1, out1);
  k_bias_elu_bf16<<<2048, BLK, 0, stream>>>(out1, b1, A2, N * H1C, H1C);

  // ---- layer 2 ----
  k_gemm_wmma<<<(Mt * (HC / 16) + 7) / 8, BLK, 0, stream>>>(A2, W2t, h2, Mt, HC / 16, H1C);
  k_att_prep<<<(N + 7) / 8, BLK, 0, stream>>>(h2, as2, ad2, als2, ald2, N, 1);
  k_edge_max<<<(Etot + BLK - 1) / BLK, BLK, 0, stream>>>(ei, E, Etot, 1, als2, ald2, m2);
  k_edge_den<<<(Etot + BLK - 1) / BLK, BLK, 0, stream>>>(ei, E, Etot, 1, als2, ald2, m2, den2);
  k_edge_agg<<<(Etot + 7) / 8, BLK, 0, stream>>>(ei, E, Etot, 1, als2, ald2, m2, den2, h2, out2);

  // ---- pool ----
  k_final_pool<<<1024, BLK, 0, stream>>>(out2, b2, batch, pool, N);
  k_pool_decode<<<(out_size + BLK - 1) / BLK, BLK, 0, stream>>>(pool, (float*)d_out, out_size);
}
